// MultiHeadAttention_64819646431588
// MI455X (gfx1250) — matrix operand order repaired and checked by reading, not compiled, not run on hardware
//
#include <hip/hip_runtime.h>

// ---------------------------------------------------------------------------
// MultiHeadAttention for MI455X (gfx1250, wave32, WMMA)
//   B=2, S=4096, D=512, H=8, dh=64
// Pipeline:
//   1) transpose_cvt:  W[n][k] fp32 -> Wt[k][n] bf16 (x4 weights).
//      Wq additionally pre-scaled by log2(e)/sqrt(dh) so QK^T logits come out
//      of the WMMA already in base-2 softmax domain (no per-logit scaling in
//      the 4096-deep key loop).
//   2) gemm_wmma<1/2/3,float>: X@W^T via v_wmma_f32_16x16x32_bf16,
//        storing qh [b,h,s,d] / khT [b,h,d,s] / vh [b,h,s,d] in bf16
//   3) flash_attn: streaming softmax, QK^T + P@V all in WMMA
//   4) gemm_wmma<0,ushort>: O @ Wo^T -> fp32 d_out
// ---------------------------------------------------------------------------

typedef __attribute__((ext_vector_type(16))) __bf16        bf16x16;
typedef __attribute__((ext_vector_type(8)))  float         f32x8;
typedef __attribute__((ext_vector_type(4)))  float         f32x4;
typedef __attribute__((ext_vector_type(4)))  unsigned int  u32x4;

union FragU {
    bf16x16        v;     // WMMA operand view
    u32x4          q[2];  // two 16B load/store views
    unsigned int   d[8];
    unsigned short s[16];
};

// round-to-nearest-even fp32 -> bf16 bits
__device__ inline unsigned short f2bf(float f) {
    unsigned int u = __float_as_uint(f);
    u += 0x7FFFu + ((u >> 16) & 1u);
    return (unsigned short)(u >> 16);
}
__device__ inline unsigned int pk2(float lo, float hi) {
    return (unsigned int)f2bf(lo) | ((unsigned int)f2bf(hi) << 16);
}

// load 8 contiguous "A" elements as packed bf16 (one fragment chunk)
__device__ inline u32x4 load8(const unsigned short* p) {
    return *(const u32x4*)p;
}
__device__ inline u32x4 load8(const float* p) {
    f32x4 a = *(const f32x4*)p;
    f32x4 b = *(const f32x4*)(p + 4);
    u32x4 r;
    r.x = pk2(a.x, a.y); r.y = pk2(a.z, a.w);
    r.z = pk2(b.x, b.y); r.w = pk2(b.z, b.w);
    return r;
}

__device__ inline f32x8 wmma_bf16(const FragU& a, const FragU& b, f32x8 c) {
    return __builtin_amdgcn_wmma_f32_16x16x32_bf16(
        /*neg_a=*/false, a.v, /*neg_b=*/false, b.v,
        /*c_mod=*/(short)0, c, /*reuse_a=*/false, /*reuse_b=*/false);
}

__device__ inline float redmax16(float v) {
#pragma unroll
    for (int m = 1; m < 16; m <<= 1) v = fmaxf(v, __shfl_xor(v, m, 32));
    return v;
}
__device__ inline float redsum16(float v) {
#pragma unroll
    for (int m = 1; m < 16; m <<= 1) v += __shfl_xor(v, m, 32);
    return v;
}

// ---------------------------------------------------------------------------
// 1) weight transpose + fp32->bf16:  Wt[k*512+n] = bf16(scale * W[n*512+k])
// ---------------------------------------------------------------------------
__global__ __launch_bounds__(256) void transpose_cvt(
    const float* __restrict__ W, unsigned short* __restrict__ Wt, float scale) {
    int idx = blockIdx.x * 256 + threadIdx.x;      // 0 .. 512*512-1
    int k = idx >> 9;
    int n = idx & 511;
    Wt[idx] = f2bf(scale * W[n * 512 + k]);
}

// ---------------------------------------------------------------------------
// 2/4) GEMM: y[M=8192,N=512] = A[M,512] @ Wt  (Wt is [K=512][N=512] bf16)
// MODE 0: fp32 row-major to outF (final projection)
// MODE 1/3: bf16 head-split [b,h,s,d]  (q / v projections)
// MODE 2:   bf16 head-split transposed [b,h,d,s]  (k projection)
// Block: 256 thr = 8 waves (4 M x 2 N); wave tile 32x64; K step 32.
// ---------------------------------------------------------------------------
template <int MODE, typename AT>
__global__ __launch_bounds__(256) void gemm_wmma_kernel(
    const AT* __restrict__ A, const unsigned short* __restrict__ Bt,
    float* __restrict__ outF, unsigned short* __restrict__ outH) {
    const int t    = threadIdx.x & 31;
    const int wave = threadIdx.x >> 5;
    const int wm   = wave >> 1;                  // 0..3
    const int wn   = wave & 1;                   // 0..1
    const int m0   = blockIdx.x * 128 + wm * 32;
    const int n0   = blockIdx.y * 128 + wn * 64;
    const int cidx = t & 15;
    const int half = t >> 4;

    f32x8 acc[2][4];
#pragma unroll
    for (int mt = 0; mt < 2; ++mt)
#pragma unroll
        for (int nt = 0; nt < 4; ++nt) acc[mt][nt] = f32x8{};

    for (int k0 = 0; k0 < 512; k0 += 32) {
        if (k0 + 32 < 512) {  // next K-slab prefetch -> global_prefetch_b8
            __builtin_prefetch(A + (m0 + cidx) * 512 + k0 + 32, 0, 1);
            __builtin_prefetch(Bt + (k0 + 32 + t) * 512 + n0, 0, 1);
        }
        FragU Af[2];
#pragma unroll
        for (int mt = 0; mt < 2; ++mt) {
            const AT* pa = A + (m0 + mt * 16 + cidx) * 512 + k0 + half * 8;
            Af[mt].q[0] = load8(pa);        // K = base .. base+7
            Af[mt].q[1] = load8(pa + 16);   // K = base+16 .. base+23
        }
        FragU Bf[4];
#pragma unroll
        for (int nt = 0; nt < 4; ++nt) {
            const unsigned short* pb = Bt + k0 * 512 + n0 + nt * 16 + cidx;
#pragma unroll
            for (int s = 0; s < 16; ++s)
                Bf[nt].s[s] = pb[(half * 8 + (s & 7) + (s >> 3) * 16) * 512];
        }
#pragma unroll
        for (int mt = 0; mt < 2; ++mt)
#pragma unroll
            for (int nt = 0; nt < 4; ++nt)
                acc[mt][nt] = wmma_bf16(Af[mt], Bf[nt], acc[mt][nt]);
    }

#pragma unroll
    for (int mt = 0; mt < 2; ++mt)
#pragma unroll
        for (int nt = 0; nt < 4; ++nt)
#pragma unroll
            for (int r = 0; r < 8; ++r) {
                const int m = m0 + mt * 16 + half * 8 + r;
                const int n = n0 + nt * 16 + cidx;
                const float val = acc[mt][nt][r];
                if (MODE == 0) {
                    outF[m * 512 + n] = val;
                } else {
                    const int b = m >> 12, s = m & 4095;
                    const int h = n >> 6,  d = n & 63;
                    if (MODE == 2)
                        outH[(((b << 3) + h) * 64 + d) * 4096 + s] = f2bf(val);
                    else
                        outH[(((b << 3) + h) * 4096 + s) * 64 + d] = f2bf(val);
                }
            }
}

// ---------------------------------------------------------------------------
// 3) Flash attention. Block = 128 thr = 4 waves; each wave owns 32 query rows.
// Streams 32 keys/iter: 8 WMMA (QK^T, contraction dh=64) + 8 WMMA (P@V).
// Logits arrive pre-scaled (scale folded into Wq), softmax in base 2.
// P layout flip (C/D -> A fragment) via wave-private padded LDS tile.
// ---------------------------------------------------------------------------
__global__ __launch_bounds__(128) void flash_attn_kernel(
    const unsigned short* __restrict__ qh,   // [b,h,s,d] bf16, pre-scaled
    const unsigned short* __restrict__ khT,  // [b,h,d,s] bf16
    const unsigned short* __restrict__ vh,   // [b,h,s,d] bf16
    unsigned short* __restrict__ Obuf) {     // [b,s,h*d] bf16
    const int t    = threadIdx.x & 31;
    const int wave = threadIdx.x >> 5;
    const int cidx = t & 15;
    const int half = t >> 4;
    const int bh   = blockIdx.y;
    const int b    = bh >> 3;
    const int h    = bh & 7;
    const int s0   = blockIdx.x * 128 + wave * 32;

    const unsigned short* Qb = qh  + (size_t)bh * 4096 * 64;
    const unsigned short* Kb = khT + (size_t)bh * 64 * 4096;
    const unsigned short* Vb = vh  + (size_t)bh * 4096 * 64;

    __shared__ unsigned short plds[4][32][40];  // pad 32->40 vs bank conflicts

    // Q fragments live in registers for the whole key loop
    FragU Qa[2][2];
#pragma unroll
    for (int mt = 0; mt < 2; ++mt)
#pragma unroll
        for (int kc = 0; kc < 2; ++kc) {
            const unsigned short* p =
                Qb + (s0 + mt * 16 + cidx) * 64 + kc * 32 + half * 8;
            Qa[mt][kc].q[0] = *(const u32x4*)p;
            Qa[mt][kc].q[1] = *(const u32x4*)(p + 16);
        }

    f32x8 acc[2][4];
    float mrow[2][8], lrow[2][8];
#pragma unroll
    for (int mt = 0; mt < 2; ++mt) {
#pragma unroll
        for (int nt = 0; nt < 4; ++nt) acc[mt][nt] = f32x8{};
#pragma unroll
        for (int r = 0; r < 8; ++r) { mrow[mt][r] = -3.0e38f; lrow[mt][r] = 0.f; }
    }

    for (int j0 = 0; j0 < 4096; j0 += 32) {
        // ---- logits S = Qs @ K^T for 32 keys (two 16-col tiles) ----
        // (softmax scale already folded into Q via Wq)
        f32x8 Sv[2][2];
#pragma unroll
        for (int jt = 0; jt < 2; ++jt) {
            FragU Bk0, Bk1;
            const unsigned short* p0 = Kb + j0 + jt * 16 + cidx;         // dh 0..31
            const unsigned short* p1 = Kb + 32 * 4096 + j0 + jt * 16 + cidx;  // dh 32..63
#pragma unroll
            for (int s = 0; s < 16; ++s) {
                const int kk = half * 8 + (s & 7) + (s >> 3) * 16;
                Bk0.s[s] = p0[kk * 4096];  Bk1.s[s] = p1[kk * 4096];
            }
#pragma unroll
            for (int mt = 0; mt < 2; ++mt) {
                f32x8 z = wmma_bf16(Qa[mt][0], Bk0, f32x8{});
                Sv[mt][jt] = wmma_bf16(Qa[mt][1], Bk1, z);
            }
        }
        // ---- streaming softmax update + write P tile (bf16) to LDS ----
#pragma unroll
        for (int mt = 0; mt < 2; ++mt) {
#pragma unroll
            for (int r = 0; r < 8; ++r) {
                const float z0 = Sv[mt][0][r];
                const float z1 = Sv[mt][1][r];
                float mx = redmax16(fmaxf(z0, z1));
                float newm = fmaxf(mrow[mt][r], mx);
                float corr = exp2f(mrow[mt][r] - newm);
                mrow[mt][r] = newm;
                float p0 = exp2f(z0 - newm);
                float p1 = exp2f(z1 - newm);
                float sum = redsum16(p0 + p1);
                lrow[mt][r] = lrow[mt][r] * corr + sum;
#pragma unroll
                for (int nt = 0; nt < 4; ++nt) acc[mt][nt][r] *= corr;
                const int row = mt * 16 + half * 8 + r;
                plds[wave][row][cidx]      = f2bf(p0);
                plds[wave][row][cidx + 16] = f2bf(p1);
            }
        }
        // ---- P as A fragments (layout flip through LDS, wave-private) ----
        FragU Pa[2];
#pragma unroll
        for (int mt = 0; mt < 2; ++mt) {
            const unsigned short* p = &plds[wave][mt * 16 + cidx][half * 8];
            Pa[mt].q[0] = *(const u32x4*)p;
            Pa[mt].q[1] = *(const u32x4*)(p + 16);
        }
        // ---- O += P @ V ----
#pragma unroll
        for (int nt = 0; nt < 4; ++nt) {
            FragU Bv;
            const unsigned short* pv = Vb + j0 * 64 + nt * 16 + cidx;
#pragma unroll
            for (int s = 0; s < 16; ++s)
                Bv.s[s] = pv[(half * 8 + (s & 7) + (s >> 3) * 16) * 64];
#pragma unroll
            for (int mt = 0; mt < 2; ++mt)
                acc[mt][nt] = wmma_bf16(Pa[mt], Bv, acc[mt][nt]);
        }
    }

    // ---- normalize and store O in [b,s,h*64+d] bf16 ----
#pragma unroll
    for (int mt = 0; mt < 2; ++mt)
#pragma unroll
        for (int r = 0; r < 8; ++r) {
            const float inv = 1.0f / lrow[mt][r];
            const int s = s0 + mt * 16 + half * 8 + r;
            unsigned short* po = Obuf + ((size_t)(b * 4096 + s)) * 512 + h * 64;
#pragma unroll
            for (int nt = 0; nt < 4; ++nt)
                po[nt * 16 + cidx] = f2bf(acc[mt][nt][r] * inv);
        }
}

// ---------------------------------------------------------------------------
// Host launcher
// ---------------------------------------------------------------------------
extern "C" void kernel_launch(void* const* d_in, const int* in_sizes, int n_in,
                              void* d_out, int out_size, void* d_ws, size_t ws_size,
                              hipStream_t stream) {
    (void)in_sizes; (void)n_in; (void)out_size; (void)ws_size;

    const float* q  = (const float*)d_in[0];
    const float* k  = (const float*)d_in[1];
    const float* v  = (const float*)d_in[2];
    const float* Wq = (const float*)d_in[3];
    const float* Wk = (const float*)d_in[4];
    const float* Wv = (const float*)d_in[5];
    const float* Wo = (const float*)d_in[6];
    float* out = (float*)d_out;

    unsigned short* ws = (unsigned short*)d_ws;
    const size_t WSZ = 512 * 512;                 // 262144 elems per weight
    const size_t HSZ = (size_t)2 * 8 * 4096 * 64; // 4194304 elems per tensor
    unsigned short* WqT = ws;
    unsigned short* WkT = WqT + WSZ;
    unsigned short* WvT = WkT + WSZ;
    unsigned short* WoT = WvT + WSZ;
    unsigned short* qh  = WoT + WSZ;
    unsigned short* khT = qh + HSZ;
    unsigned short* vhB = khT + HSZ;
    unsigned short* Ob  = vhB + HSZ;              // total ~34 MB of d_ws

    // softmax scale (1/sqrt(dh)) and base-2 conversion folded into Wq
    const float QSCALE = 1.4426950408889634f * 0.125f;
    transpose_cvt<<<1024, 256, 0, stream>>>(Wq, WqT, QSCALE);
    transpose_cvt<<<1024, 256, 0, stream>>>(Wk, WkT, 1.0f);
    transpose_cvt<<<1024, 256, 0, stream>>>(Wv, WvT, 1.0f);
    transpose_cvt<<<1024, 256, 0, stream>>>(Wo, WoT, 1.0f);

    dim3 ggrid(64, 4);  // M=8192/128, N=512/128
    gemm_wmma_kernel<1, float><<<ggrid, 256, 0, stream>>>(q, WqT, nullptr, qh);
    gemm_wmma_kernel<2, float><<<ggrid, 256, 0, stream>>>(k, WkT, nullptr, khT);
    gemm_wmma_kernel<3, float><<<ggrid, 256, 0, stream>>>(v, WvT, nullptr, vhB);

    flash_attn_kernel<<<dim3(32, 16), 128, 0, stream>>>(qh, khT, vhB, Ob);

    gemm_wmma_kernel<0, unsigned short><<<ggrid, 256, 0, stream>>>(Ob, WoT, out, nullptr);
}
